// LlamaPEER_25305947308157
// MI455X (gfx1250) — matrix operand order repaired and checked by reading, not compiled, not run on hardware
//
#include <hip/hip_runtime.h>
#include <hip/hip_bf16.h>
#include <math.h>

// ---------------------------------------------------------------------------
// LlamaPEER forward for MI455X (gfx1250, wave32)
//   B=2 T=1024 D=2048 H=4 K=8 DK=64 E=16384 NK=128
// Memory-bound on expert-row gathers (~1.07 GB/fwd -> ~46us @ 23.3 TB/s).
//   k1: q = x @ Wq via v_wmma_f32_16x16x32_f16 (f32 accumulate),
//       fragment-packed LDS tiles so each lane builds A/B with ds_load_b128.
//   k2: fused sims + product-key top-k.
//   k3: expert gather/GEMV, double-buffered global_load_async_to_lds_b128.
// ---------------------------------------------------------------------------

typedef __attribute__((ext_vector_type(16))) _Float16 v16h;
typedef __attribute__((ext_vector_type(8)))  float    v8f;

// async-to-LDS builtin is typed as pointers to 16B int vectors
typedef int peer_v4i __attribute__((vector_size(16)));
typedef __attribute__((address_space(1))) peer_v4i* peer_gp;
typedef __attribute__((address_space(3))) peer_v4i* peer_lp;

constexpr int Bc  = 2;
constexpr int Tc  = 1024;
constexpr int Dc  = 2048;
constexpr int Hc  = 4;
constexpr int Kc  = 8;
constexpr int DKc = 64;
constexpr int NKc = 128;
constexpr int Mq  = Bc * Tc;        // 2048 rows
constexpr int Nq  = 2 * Hc * DKc;   // 512 cols
constexpr float NEG_BIG = -3.0e38f;

#if __has_builtin(__builtin_amdgcn_global_load_async_to_lds_b128) && \
    __has_builtin(__builtin_amdgcn_s_wait_asynccnt)
#define PEER_ASYNC 1
#else
#define PEER_ASYNC 0
#endif

// ---------------------------------------------------------------------------
// Kernel 1: q = x @ Wq  (2048x2048 @ 2048x512), WMMA f16 inputs / f32 acc.
// Block = 256 threads (8 waves). Block tile = 16 rows x 256 cols; each wave
// owns two adjacent 16x16 tiles (reuses its A fragment). K steps of 32.
// LDS layouts are padded to 80B row stride: 16B-aligned fragment chunks, and
// bank-spread reads, so fragment builds lower to ds_load_b128.
// ---------------------------------------------------------------------------
__global__ __launch_bounds__(256) void qproj_wmma(const float* __restrict__ x,
                                                  const float* __restrict__ Wq,
                                                  float* __restrict__ q) {
    __shared__ _Float16 As[16][40];    // row-major A tile, 32 K used + pad
    __shared__ _Float16 Bt[256][40];   // COLUMN-major B tile: Bt[col][k]

    const int tid  = threadIdx.x;
    const int lane = tid & 31;
    const int wave = tid >> 5;               // 0..7
    const int rowBase = blockIdx.x * 16;
    const int colBase = blockIdx.y * 256;

    v8f c0 = {};
    v8f c1 = {};

    for (int kk = 0; kk < Dc; kk += 32) {
        // Stage A tile: 16x32, 2 elements per thread, coalesced over k
        {
            int i0 = tid * 2;
            int r  = i0 >> 5;
            int cc = i0 & 31;
            const float* xr = x + (size_t)(rowBase + r) * Dc + kk;
            As[r][cc]     = (_Float16)xr[cc];
            As[r][cc + 1] = (_Float16)xr[cc + 1];
        }
        // Stage B tile: 32(k) x 256(col), coalesced global reads over cols,
        // transposed into column-major LDS so lane reads are contiguous.
        #pragma unroll
        for (int i = 0; i < 32; ++i) {
            int idx = tid + i * 256;
            int r   = idx >> 8;      // k row 0..31
            int cc  = idx & 255;     // col
            Bt[cc][r] = (_Float16)Wq[(size_t)(kk + r) * Nq + colBase + cc];
        }
        __syncthreads();

        // A fragment (16x32 f16): lane(row=lane%16, half=lane/16):
        //   element e -> K = half*16+e (contiguous 32B run)
        const int arow  = lane & 15;
        const int ahalf = lane >> 4;
        v16h a, b0, b1;
        #pragma unroll
        for (int e = 0; e < 16; ++e) {
            a[e]     = As[arow][ahalf * 16 + e];
        }
        // B fragments (32x16 f16 each): lane(col=base+lane%16, half=lane/16),
        // element e -> K = half*16+e : one contiguous 32B run per lane.
        const int bcol0 = wave * 32 + (lane & 15);
        const int bcol1 = bcol0 + 16;
        #pragma unroll
        for (int e = 0; e < 16; ++e) {
            b0[e] = Bt[bcol0][ahalf * 16 + e];
            b1[e] = Bt[bcol1][ahalf * 16 + e];
        }

        c0 = __builtin_amdgcn_wmma_f32_16x16x32_f16(false, a, false, b0,
                                                    (short)0, c0, false, false);
        c1 = __builtin_amdgcn_wmma_f32_16x16x32_f16(false, a, false, b1,
                                                    (short)0, c1, false, false);
        __syncthreads();
    }

    // C/D layout: VGPR r -> M = r + 8*(lane>=16), N = lane%16
    const int mbase = rowBase + ((lane >> 4) << 3);
    const int ncol0 = colBase + wave * 32 + (lane & 15);
    #pragma unroll
    for (int r = 0; r < 8; ++r) {
        q[(size_t)(mbase + r) * Nq + ncol0]      = c0[r];
        q[(size_t)(mbase + r) * Nq + ncol0 + 16] = c1[r];
    }
}

// ---------------------------------------------------------------------------
// Kernel 2: per (b,t,h): sims vs 128+128 product keys, top-8 each half,
// 64 pairwise sums, top-8 of those -> final scores + expert indices.
// One 128-thread block per (b,t,h).
// ---------------------------------------------------------------------------
__global__ __launch_bounds__(128) void sim_topk(const float* __restrict__ q,
                                                const float* __restrict__ keys,
                                                float* __restrict__ out_scores,
                                                int* __restrict__ out_idx) {
    const int bt  = blockIdx.x / Hc;
    const int h   = blockIdx.x % Hc;
    const int tid = threadIdx.x;

    __shared__ float q1[DKc], q2[DKc];
    __shared__ float sim1[NKc], sim2[NKc];
    __shared__ float s1[Kc], s2[Kc];
    __shared__ int   i1[Kc], i2[Kc];
    __shared__ float comb[64];
    __shared__ int   combi[64];
    __shared__ float red[128];
    __shared__ int   redi[128];

    const float* qrow = q + (size_t)bt * Nq + h * 2 * DKc;
    if (tid < 64) q1[tid]      = qrow[tid];
    else          q2[tid - 64] = qrow[tid];
    __syncthreads();

    {
        const float* k1 = keys + (((size_t)h * NKc + tid) * 2 + 0) * DKc;
        const float* k2 = k1 + DKc;
        float a1 = 0.f, a2 = 0.f;
        #pragma unroll 8
        for (int d = 0; d < DKc; ++d) {
            a1 = fmaf(q1[d], k1[d], a1);
            a2 = fmaf(q2[d], k2[d], a2);
        }
        sim1[tid] = a1;
        sim2[tid] = a2;
    }
    __syncthreads();

    for (int half = 0; half < 2; ++half) {
        float* sim = half ? sim2 : sim1;
        float* sv  = half ? s2 : s1;
        int*   si  = half ? i2 : i1;
        for (int it = 0; it < Kc; ++it) {
            red[tid]  = sim[tid];
            redi[tid] = tid;
            __syncthreads();
            for (int s = 64; s > 0; s >>= 1) {
                if (tid < s && red[tid + s] > red[tid]) {
                    red[tid]  = red[tid + s];
                    redi[tid] = redi[tid + s];
                }
                __syncthreads();
            }
            if (tid == 0) {
                int sl = redi[0];
                sv[it] = red[0];
                si[it] = sl;
                sim[sl] = NEG_BIG;
            }
            __syncthreads();
        }
    }

    if (tid < 64) {
        comb[tid]  = s1[tid >> 3] + s2[tid & 7];
        combi[tid] = i1[tid >> 3] * NKc + i2[tid & 7];
    }
    __syncthreads();

    for (int it = 0; it < Kc; ++it) {
        red[tid]  = (tid < 64) ? comb[tid] : NEG_BIG;
        redi[tid] = (tid < 64) ? tid : 0;
        __syncthreads();
        for (int s = 64; s > 0; s >>= 1) {
            if (tid < s && red[tid + s] > red[tid]) {
                red[tid]  = red[tid + s];
                redi[tid] = redi[tid + s];
            }
            __syncthreads();
        }
        if (tid == 0) {
            int sl = redi[0];
            out_scores[(size_t)blockIdx.x * Kc + it] = red[0];
            out_idx[(size_t)blockIdx.x * Kc + it]    = combi[sl];
            comb[sl] = NEG_BIG;
        }
        __syncthreads();
    }
}

// ---------------------------------------------------------------------------
// Kernel 3: per token: gather 32 expert rows (down+up), hidden = silu(x.wd)
// * relu(score), out += hidden * wu. One 256-thread block per (b,t).
// Async path: double-buffered global_load_async_to_lds_b128; each thread
// streams 32B of each 8KB row (2 b128 chunks), 4 async instr per expert,
// s_wait_asynccnt 4 overlaps expert i+1's DMA with expert i's compute.
// ---------------------------------------------------------------------------
__global__ __launch_bounds__(256) void expert_moe(const float* __restrict__ x,
                                                  const float* __restrict__ e_down,
                                                  const float* __restrict__ e_up,
                                                  const float* __restrict__ scores,
                                                  const int* __restrict__ idxs,
                                                  float* __restrict__ out) {
    const int bt  = blockIdx.x;
    const int tid = threadIdx.x;

    __shared__ float xs[Dc];
    __shared__ float red[8];

    #pragma unroll
    for (int i = 0; i < Dc / 256; ++i)
        xs[tid + i * 256] = x[(size_t)bt * Dc + tid + i * 256];

    float acc[8] = {0.f, 0.f, 0.f, 0.f, 0.f, 0.f, 0.f, 0.f};
    const int d0 = tid * 8;   // thread owns d0..d0+7
    const int idx_base = bt * (Hc * Kc);

#if PEER_ASYNC
    __shared__ float wds[2][Dc];   // down-row double buffer (16KB)
    __shared__ float wus[2][Dc];   // up-row double buffer (16KB)

    #define PEER_ASYNC_FETCH(eidx, buf)                                        \
        do {                                                                   \
            const float* wdn_ = e_down + (size_t)(eidx) * Dc;                  \
            const float* wun_ = e_up   + (size_t)(eidx) * Dc;                  \
            __builtin_amdgcn_global_load_async_to_lds_b128(                    \
                (peer_gp)(void*)(wdn_ + d0),                                   \
                (peer_lp)(void*)&wds[buf][d0], 0, 0);                          \
            __builtin_amdgcn_global_load_async_to_lds_b128(                    \
                (peer_gp)(void*)(wdn_ + d0 + 4),                               \
                (peer_lp)(void*)&wds[buf][d0 + 4], 0, 0);                      \
            __builtin_amdgcn_global_load_async_to_lds_b128(                    \
                (peer_gp)(void*)(wun_ + d0),                                   \
                (peer_lp)(void*)&wus[buf][d0], 0, 0);                          \
            __builtin_amdgcn_global_load_async_to_lds_b128(                    \
                (peer_gp)(void*)(wun_ + d0 + 4),                               \
                (peer_lp)(void*)&wus[buf][d0 + 4], 0, 0);                      \
        } while (0)

    PEER_ASYNC_FETCH(idxs[idx_base], 0);   // prologue: expert 0 -> buf 0
    __syncthreads();                        // xs visible to all waves

    for (int hk = 0; hk < Hc * Kc; ++hk) {
        const int cur = hk & 1;
        if (hk + 1 < Hc * Kc) {
            PEER_ASYNC_FETCH(idxs[idx_base + hk + 1], cur ^ 1);
            __builtin_amdgcn_s_wait_asynccnt(4);  // cur's 4 chunks resident
        } else {
            __builtin_amdgcn_s_wait_asynccnt(0);
        }
        __syncthreads();   // all waves' slices of buf[cur] resident

        const float sc = scores[idx_base + hk];
        float p = 0.f;
        #pragma unroll
        for (int i = 0; i < 8; ++i)
            p = fmaf(xs[d0 + i], wds[cur][d0 + i], p);

        #pragma unroll
        for (int off = 16; off > 0; off >>= 1)
            p += __shfl_xor(p, off);
        if ((tid & 31) == 0) red[tid >> 5] = p;
        __syncthreads();
        if (tid == 0) {
            float d = 0.f;
            #pragma unroll
            for (int w = 0; w < 8; ++w) d += red[w];
            red[0] = d;
        }
        __syncthreads();
        const float dot = red[0];

        const float hidden = (dot / (1.f + __expf(-dot))) * fmaxf(sc, 0.f);
        #pragma unroll
        for (int i = 0; i < 8; ++i)
            acc[i] = fmaf(hidden, wus[cur][d0 + i], acc[i]);
        __syncthreads();   // done with buf[cur] & red before reuse
    }
    #undef PEER_ASYNC_FETCH
#else
    __syncthreads();
    for (int hk = 0; hk < Hc * Kc; ++hk) {
        const int   e  = idxs[idx_base + hk];
        const float sc = scores[idx_base + hk];
        const float* wd = e_down + (size_t)e * Dc;
        const float* wu = e_up   + (size_t)e * Dc;

        if (hk + 1 < Hc * Kc) {
            int en = idxs[idx_base + hk + 1];
            __builtin_prefetch(e_down + (size_t)en * Dc + d0, 0, 1);
            __builtin_prefetch(e_up   + (size_t)en * Dc + d0, 0, 1);
        }

        const float4* wd4 = (const float4*)wd + tid * 2;
        float4 v0 = wd4[0], v1 = wd4[1];
        float p = xs[d0 + 0] * v0.x + xs[d0 + 1] * v0.y +
                  xs[d0 + 2] * v0.z + xs[d0 + 3] * v0.w +
                  xs[d0 + 4] * v1.x + xs[d0 + 5] * v1.y +
                  xs[d0 + 6] * v1.z + xs[d0 + 7] * v1.w;

        #pragma unroll
        for (int off = 16; off > 0; off >>= 1)
            p += __shfl_xor(p, off);
        if ((tid & 31) == 0) red[tid >> 5] = p;
        __syncthreads();
        if (tid == 0) {
            float d = 0.f;
            #pragma unroll
            for (int w = 0; w < 8; ++w) d += red[w];
            red[0] = d;
        }
        __syncthreads();
        const float dot = red[0];
        __syncthreads();

        const float hidden = (dot / (1.f + __expf(-dot))) * fmaxf(sc, 0.f);
        const float4* wu4 = (const float4*)wu + tid * 2;
        float4 u0 = wu4[0], u1 = wu4[1];
        acc[0] = fmaf(hidden, u0.x, acc[0]);
        acc[1] = fmaf(hidden, u0.y, acc[1]);
        acc[2] = fmaf(hidden, u0.z, acc[2]);
        acc[3] = fmaf(hidden, u0.w, acc[3]);
        acc[4] = fmaf(hidden, u1.x, acc[4]);
        acc[5] = fmaf(hidden, u1.y, acc[5]);
        acc[6] = fmaf(hidden, u1.z, acc[6]);
        acc[7] = fmaf(hidden, u1.w, acc[7]);
    }
#endif

    float4* o = (float4*)(out + (size_t)bt * Dc);
    o[tid * 2]     = make_float4(acc[0], acc[1], acc[2], acc[3]);
    o[tid * 2 + 1] = make_float4(acc[4], acc[5], acc[6], acc[7]);
}

// ---------------------------------------------------------------------------
extern "C" void kernel_launch(void* const* d_in, const int* in_sizes, int n_in,
                              void* d_out, int out_size, void* d_ws, size_t ws_size,
                              hipStream_t stream) {
    const float* x      = (const float*)d_in[0];  // (B,T,D)
    const float* Wq     = (const float*)d_in[1];  // (D, 2*H*DK)
    const float* keys   = (const float*)d_in[2];  // (H,NK,2,DK)
    const float* e_down = (const float*)d_in[3];  // (E,D)
    const float* e_up   = (const float*)d_in[4];  // (E,D)
    float* out = (float*)d_out;                   // (B,T,D)

    float* qW      = (float*)d_ws;                         // Mq*Nq floats
    float* scoresW = qW + (size_t)Mq * Nq;                 // Mq*H*K floats
    int*   idxW    = (int*)(scoresW + (size_t)Mq * Hc * Kc);

    dim3 g1(Mq / 16, Nq / 256);
    qproj_wmma<<<g1, 256, 0, stream>>>(x, Wq, qW);

    sim_topk<<<Mq * Hc, 128, 0, stream>>>(qW, keys, scoresW, idxW);

    expert_moe<<<Mq, 256, 0, stream>>>(x, e_down, e_up, scoresW, idxW, out);
}
